// VanillaNeRF_11991548690836
// MI455X (gfx1250) — matrix operand order repaired and checked by reading, not compiled, not run on hardware
//
#include <hip/hip_runtime.h>

typedef __attribute__((ext_vector_type(16))) _Float16 v16h;
typedef __attribute__((ext_vector_type(8)))  float    v8f;

// NeRF frequency bands 2^linspace(0,10,4), pre-scaled by 1/(2*pi) so we can
// feed v_sin_f32 / v_cos_f32 (which take input in turns) directly.
#define SB0 0.15915494f      // 1.0      / 2pi
#define SB1 1.6041813f       // 2^(10/3) / 2pi
#define SB2 16.1691461f      // 2^(20/3) / 2pi
#define SB3 162.9746617f     // 2^10     / 2pi

// Single-instruction ReLU: v_med3_f32(x, 0, +inf)
__device__ __forceinline__ float relu1(float v) {
    return __builtin_amdgcn_fmed3f(v, 0.0f, __builtin_inff());
}

// Swap the two 16-lane halves of a wave32 (v_permlanex16_b32, identity selects).
__device__ __forceinline__ float swap_half16(float v) {
    int i = __builtin_bit_cast(int, v);
    int r = __builtin_amdgcn_permlanex16(i, i, 0x76543210, 0xfedcba98, false, false);
    return __builtin_bit_cast(float, r);
}

// This half-wave's B-fragment column: 16 encoded features of one point.
// lo half -> f[0..15], hi half -> f[16..31]; f[27..30]=0, f[31]=1 (bias row).
__device__ __forceinline__ v16h encode_col(float px, float py, float pz, bool hi) {
    const float bx0 = px * SB0, by0 = py * SB0, bz0 = pz * SB0;
    const float bx1 = px * SB1, by1 = py * SB1, bz1 = pz * SB1;
    const float bx2 = px * SB2, by2 = py * SB2, bz2 = pz * SB2;
    const float bx3 = px * SB3, by3 = py * SB3, bz3 = pz * SB3;

    float sA[7], cA[6];
    sA[0] = hi ? by2 : bx0;  sA[1] = hi ? bz2 : by0;  sA[2] = hi ? bx3 : bz0;
    sA[3] = hi ? by3 : bx1;  sA[4] = hi ? bz3 : by1;  sA[5] = hi ? bz3 : bz1;
    sA[6] = hi ? bz3 : bx2;
    cA[0] = hi ? bx2 : bx0;  cA[1] = hi ? by2 : by0;  cA[2] = hi ? bz2 : bz0;
    cA[3] = hi ? bx3 : bx1;  cA[4] = hi ? by3 : by1;  cA[5] = hi ? bz3 : bz1;

    float S[7], C[6];
#pragma unroll
    for (int i = 0; i < 7; ++i) S[i] = __builtin_amdgcn_sinf(sA[i]);
#pragma unroll
    for (int i = 0; i < 6; ++i) C[i] = __builtin_amdgcn_cosf(cA[i]);

    float fl[16];
    fl[0]  = hi ? S[0] : px;   fl[1]  = hi ? S[1] : py;
    fl[2]  = hi ? C[0] : pz;   fl[3]  = hi ? C[1] : S[0];
    fl[4]  = hi ? C[2] : S[1]; fl[5]  = S[2];
    fl[6]  = hi ? S[3] : C[0]; fl[7]  = hi ? S[4] : C[1];
    fl[8]  = hi ? C[3] : C[2]; fl[9]  = hi ? C[4] : S[3];
    fl[10] = hi ? C[5] : S[4]; fl[11] = hi ? 0.0f : S[5];
    fl[12] = hi ? 0.0f : C[3]; fl[13] = hi ? 0.0f : C[4];
    fl[14] = hi ? 0.0f : C[5]; fl[15] = hi ? 1.0f : S[6];   // 1.0 -> bias row K=31

    v16h bf;
#pragma unroll
    for (int i = 0; i < 16; ++i) bf[i] = (_Float16)fl[i];
    return bf;
}

// Layer-2 in-register reduction (16 hidden values per lane), 4 fma chains.
__device__ __forceinline__ float layer2(const v8f& c0, const v8f& c1,
                                        const float* w20, const float* w21) {
    float a0 = 0.f, a1 = 0.f, a2 = 0.f, a3 = 0.f;
#pragma unroll
    for (int r = 0; r < 8; r += 2) {
        a0 = fmaf(relu1(c0[r]),     w20[r],     a0);
        a1 = fmaf(relu1(c0[r + 1]), w20[r + 1], a1);
        a2 = fmaf(relu1(c1[r]),     w21[r],     a2);
        a3 = fmaf(relu1(c1[r + 1]), w21[r + 1], a3);
    }
    return (a0 + a1) + (a2 + a3);
}

// Transposed fused NeRF MLP:
//   D(32x16) = [W1;b1]^T(32x32) * [enc;1]^T(32x16) via 2x v_wmma_f32_16x16x32_f16
//   A = weights+bias (loop-invariant), C = inline 0, D columns = points.
//   Two tiles per iteration, fully straight-line (only stores are predicated)
//   so the scheduler can interleave loads/encodes/WMMAs across both tiles.
__global__ void __launch_bounds__(256)
nerf_fused_wmma(const float* __restrict__ x,
                const float* __restrict__ W1,   // [27,32] row-major
                const float* __restrict__ b1,   // [32]
                const float* __restrict__ W2,   // [32,1]
                const float* __restrict__ b2,   // [1]
                float* __restrict__ out,        // [N]
                int N)
{
    const int  lane = threadIdx.x & 31;
    const bool hi   = (lane & 16) != 0;
    const int  n0   = lane & 15;          // fragment column (= point within tile)
    const int  K0   = hi ? 16 : 0;        // A-fragment K offset for this half
    const int  mB   = hi ? 8 : 0;         // C/D row offset for this half

    // A fragments = W1^T rows (hidden units), K = features; K=31 carries b1.
    v16h a0, a1;
#pragma unroll
    for (int i = 0; i < 8; ++i) {
        const int kA = K0 + i;            // < 24, always a real weight
        const int kB = K0 + 8 + i;        // up to 31
        a0[i] = (_Float16)W1[kA * 32 + n0];
        a1[i] = (_Float16)W1[kA * 32 + 16 + n0];
        float wb0, wb1;
        if (kB < 27)       { wb0 = W1[kB * 32 + n0]; wb1 = W1[kB * 32 + 16 + n0]; }
        else if (kB == 31) { wb0 = b1[n0];           wb1 = b1[16 + n0]; }
        else               { wb0 = 0.0f;             wb1 = 0.0f; }
        a0[8 + i] = (_Float16)wb0;
        a1[8 + i] = (_Float16)wb1;
    }

    // Per-row layer-2 weights for this half (C/D row M = mB + r).
    float w20[8], w21[8];
#pragma unroll
    for (int r = 0; r < 8; ++r) {
        w20[r] = W2[mB + r];
        w21[r] = W2[16 + mB + r];
    }
    const float bias2 = b2[0];
    const v8f   czero = {};               // selects the inline SRC2=0 form

    // Wave/tile bookkeeping in SGPRs (scalar loop, scalar compares).
    const int numTiles      = N >> 4;
    const int wavesPerBlock = blockDim.x >> 5;
    const int waveInBlk = __builtin_amdgcn_readfirstlane(threadIdx.x >> 5);
    const int wave      = blockIdx.x * wavesPerBlock + waveInBlk;
    const int wstride   = gridDim.x * wavesPerBlock;
    const int wstride2  = 2 * wstride;
    const float3* __restrict__ xp = (const float3*)x;

    for (int t0 = wave; t0 < numTiles; t0 += wstride2) {
        const int  t1    = t0 + wstride;        // scalar
        const bool haveB = (t1 < numTiles);     // scalar
        const int  tB    = haveB ? t1 : t0;     // clamped -> always safe
        const int  nxt   = t0 + wstride2;
        if (nxt < numTiles)
            __builtin_prefetch(&xp[(nxt << 4) + n0], 0, 1);

        // Both coordinate loads issued up front (global_load_b96 x2);
        // tile-0's encode hides tile-1's load latency.
        const float3 pA = xp[(t0 << 4) + n0];
        const float3 pB = xp[(tB << 4) + n0];

        // ---- tile 0 ----
        const v16h bfA = encode_col(pA.x, pA.y, pA.z, hi);
        v8f cA0 = __builtin_amdgcn_wmma_f32_16x16x32_f16(
                      false, a0, false, bfA, (short)0, czero, false, false);
        v8f cA1 = __builtin_amdgcn_wmma_f32_16x16x32_f16(
                      false, a1, false, bfA, (short)0, czero, false, false);

        // ---- tile 1 (unconditional compute; only its store is predicated) ----
        const v16h bfB = encode_col(pB.x, pB.y, pB.z, hi);
        v8f cB0 = __builtin_amdgcn_wmma_f32_16x16x32_f16(
                      false, a0, false, bfB, (short)0, czero, false, false);
        v8f cB1 = __builtin_amdgcn_wmma_f32_16x16x32_f16(
                      false, a1, false, bfB, (short)0, czero, false, false);

        // ---- epilogues ----
        float accA = layer2(cA0, cA1, w20, w21);
        accA += swap_half16(accA);
        if (!hi)
            out[(t0 << 4) + n0] = relu1(accA + bias2);   // 64B coalesced

        float accB = layer2(cB0, cB1, w20, w21);
        accB += swap_half16(accB);
        if (haveB && !hi)
            out[(t1 << 4) + n0] = relu1(accB + bias2);
    }
}

extern "C" void kernel_launch(void* const* d_in, const int* in_sizes, int n_in,
                              void* d_out, int out_size, void* d_ws, size_t ws_size,
                              hipStream_t stream)
{
    const float* x  = (const float*)d_in[0];
    const float* W1 = (const float*)d_in[1];
    const float* b1 = (const float*)d_in[2];
    const float* W2 = (const float*)d_in[3];
    const float* b2 = (const float*)d_in[4];
    float* out = (float*)d_out;

    const int N = in_sizes[0] / 3;          // 2,097,152 points
    const int tiles = N >> 4;
    const int threads = 256;                // 8 waves/block (wave32)
    const int wavesPerBlock = threads / 32;
    int blocks = (tiles + wavesPerBlock - 1) / wavesPerBlock;
    if (blocks > 2048) blocks = 2048;
    if (blocks < 1) blocks = 1;

    nerf_fused_wmma<<<blocks, threads, 0, stream>>>(x, W1, b1, W2, b2, out, N);
}
